// SelfAttention_15925738734108
// MI455X (gfx1250) — matrix operand order repaired and checked by reading, not compiled, not run on hardware
//
#include <hip/hip_runtime.h>

#define DEV __device__ __forceinline__

typedef __attribute__((ext_vector_type(16))) __bf16 v16bf;
typedef __attribute__((ext_vector_type(8)))  float  v8f;

union FragBF { v16bf v; unsigned int u[8]; };

// ---- float pair -> packed bf16 (hardware cvt) ------------------------------
DEV unsigned int cvt_pk_bf16(float lo, float hi) {
#if __has_builtin(__builtin_amdgcn_cvt_pk_bf16_f32)
  typedef __attribute__((ext_vector_type(2))) __bf16 v2bf;
  union { v2bf v; unsigned int u; } c;
  c.v = __builtin_amdgcn_cvt_pk_bf16_f32(lo, hi);
  return c.u;
#else
  unsigned int r;
  asm("v_cvt_pk_bf16_f32 %0, %1, %2" : "=v"(r) : "v"(lo), "v"(hi));
  return r;
#endif
}

static constexpr int B_ = 8, C_ = 512, L_ = 2048, D_ = 64;

// ---- vectorized fragment loaders (word = u32 = bf16 pair) ------------------
// 16-bit A 16x32 layout: lane (h,ln) VGPR r holds k = 16*(r>>2)+2*(r&3)+8h.
// With row-major bf16 rows those 8 words are [base+4h+0..3] and [base+4h+8..11].
DEV void loadA16x32(FragBF& A, const unsigned int* p) {   // p = rowbase + 4*h + off
  uint4 lo = *(const uint4*)(p);
  uint4 hi = *(const uint4*)(p + 8);
  A.u[0] = lo.x; A.u[1] = lo.y; A.u[2] = lo.z; A.u[3] = lo.w;
  A.u[4] = hi.x; A.u[5] = hi.y; A.u[6] = hi.z; A.u[7] = hi.w;
}
// B 32x16 loader, contiguous k order: lane (h,n) VGPR r holds k = 2r+16h -> words n*RS+8h+{0..7}.
DEV void loadB32x16(FragBF& Bf, const unsigned int* p) {  // p = colbase + 8*h + off
  uint4 lo = *(const uint4*)(p);
  uint4 hi = *(const uint4*)(p + 4);
  Bf.u[0] = lo.x; Bf.u[1] = lo.y; Bf.u[2] = lo.z; Bf.u[3] = lo.w;
  Bf.u[4] = hi.x; Bf.u[5] = hi.y; Bf.u[6] = hi.z; Bf.u[7] = hi.w;
}

// ---------------------------------------------------------------------------
// Projection: Y[b] = W(MxC) @ x[b](CxL) + bias, output bf16.
// TR=1: store Y^T as [b][l][64]  (Q and K^T: pairs contiguous in d)
// TR=0: store Y   as [b][m][L]   (V: pairs contiguous in l)
// ---------------------------------------------------------------------------
template<int TR>
__global__ __launch_bounds__(256)
void proj_bf16_kernel(const float* __restrict__ x, const float* __restrict__ W,
                      const float* __restrict__ bias, unsigned short* __restrict__ y,
                      int M, int Mt)
{
  const int lane = threadIdx.x & 31;
  const int h = lane >> 4, ln = lane & 15;
  const int gw = blockIdx.x * 8 + (threadIdx.x >> 5);
  const int nt = gw & 127;               // L/16 = 128 n-tiles
  const int mt = (gw >> 7) % Mt;
  const int b  = gw / (128 * Mt);
  const int m0 = mt * 16, n0 = nt * 16;
  const float* xb = x + (size_t)b * C_ * L_;

  v8f acc = {};
  for (int k0 = 0; k0 < C_; k0 += 32) {
    FragBF A, Bf;
    // A = W tile: f32 row, convert. Pairs at k = {0,2,4,6}+8h and {16,..}+8h.
    const float* wrow = W + (size_t)(m0 + ln) * C_ + k0 + 8 * h;
    float4 a0 = *(const float4*)(wrow);
    float4 a1 = *(const float4*)(wrow + 4);
    float4 a2 = *(const float4*)(wrow + 16);
    float4 a3 = *(const float4*)(wrow + 20);
    A.u[0] = cvt_pk_bf16(a0.x, a0.y); A.u[1] = cvt_pk_bf16(a0.z, a0.w);
    A.u[2] = cvt_pk_bf16(a1.x, a1.y); A.u[3] = cvt_pk_bf16(a1.z, a1.w);
    A.u[4] = cvt_pk_bf16(a2.x, a2.y); A.u[5] = cvt_pk_bf16(a2.z, a2.w);
    A.u[6] = cvt_pk_bf16(a3.x, a3.y); A.u[7] = cvt_pk_bf16(a3.z, a3.w);
    // B = x tile: k = channel (stride L), n = position
#pragma unroll
    for (int r = 0; r < 8; ++r) {
      const int kb = k0 + 2 * (r & 3) + 8 * h + 16 * (r >> 2);
      const float e0 = xb[(size_t)kb * L_ + n0 + ln];
      const float e1 = xb[(size_t)(kb + 1) * L_ + n0 + ln];
      Bf.u[r] = cvt_pk_bf16(e0, e1);
    }
    acc = __builtin_amdgcn_wmma_f32_16x16x32_bf16(false, A.v, false, Bf.v,
                                                  (short)0, acc, false, false);
  }
  // epilogue: bias + bf16 store. C/D layout: element e -> m = m0 + e + 8h.
  const float* bp = bias + m0 + 8 * h;
  float4 b0 = *(const float4*)(bp);
  float4 b1 = *(const float4*)(bp + 4);
  float vals[8] = { acc[0] + b0.x, acc[1] + b0.y, acc[2] + b0.z, acc[3] + b0.w,
                    acc[4] + b1.x, acc[5] + b1.y, acc[6] + b1.z, acc[7] + b1.w };
  if (TR) {
    unsigned int* yw = (unsigned int*)y;
#pragma unroll
    for (int e = 0; e < 8; e += 2) {
      const size_t widx = (((size_t)b * L_ + n0 + ln) * D_ + m0 + 8 * h + e) >> 1;
      yw[widx] = cvt_pk_bf16(vals[e], vals[e + 1]);
    }
  } else {
#pragma unroll
    for (int e = 0; e < 8; ++e) {
      const int m = m0 + e + 8 * h;
      y[((size_t)b * M + m) * L_ + n0 + ln] = (unsigned short)cvt_pk_bf16(vals[e], vals[e]);
    }
  }
}

// ---------------------------------------------------------------------------
// Flash attention: one block per (b, 32-query tile). 8 waves; wave w owns
// output channels [64w, 64w+64). Two 16-query subtiles share V A-fragments.
// Computes S^T = K @ Q^T and O^T = V^T @ P^T so softmax rows are per-lane
// scalars (row = lane&15).
// ---------------------------------------------------------------------------
__global__ __launch_bounds__(256)
void attn_kernel(const unsigned short* __restrict__ qbuf,  // [B][L][64] bf16
                 const unsigned short* __restrict__ kbuf,  // [B][L][64] bf16 (= k^T)
                 const unsigned short* __restrict__ vbuf,  // [B][C][L]  bf16
                 const float* __restrict__ x,
                 const float* __restrict__ gamma_p,
                 float* __restrict__ out)
{
  const int lane = threadIdx.x & 31;
  const int h = lane >> 4, ln = lane & 15;
  const int w = threadIdx.x >> 5;
  const int b  = blockIdx.x >> 6;            // 64 query tiles of 32 per batch
  const int l0 = (blockIdx.x & 63) * 32;
  const float gamma = gamma_p[0];

  const unsigned int* q32 = (const unsigned int*)qbuf + (size_t)b * L_ * 32;
  const unsigned int* k32 = (const unsigned int*)kbuf + (size_t)b * L_ * 32;
  const unsigned int* v32 = (const unsigned int*)vbuf + (size_t)b * C_ * 1024;

  // Q^T B-operand fragments: [query subtile][d-half]
  FragBF qb[2][2];
#pragma unroll
  for (int q = 0; q < 2; ++q)
#pragma unroll
    for (int kh = 0; kh < 2; ++kh)
      loadA16x32(qb[q][kh], q32 + (size_t)(l0 + 16 * q + ln) * 32 + 4 * h + 16 * kh);

  v8f zero = {};
  v8f acc[2][4];
#pragma unroll
  for (int q = 0; q < 2; ++q)
#pragma unroll
    for (int t = 0; t < 4; ++t) acc[q][t] = zero;
  float m_run[2] = { -1e30f, -1e30f };
  float l_run[2] = { 0.0f, 0.0f };

  for (int j0 = 0; j0 < L_; j0 += 32) {
    // ---- K A-fragments for this key tile: [j subtile][d-half] -------------
    FragBF Ka[2][2];
#pragma unroll
    for (int jt = 0; jt < 2; ++jt)
#pragma unroll
      for (int kh = 0; kh < 2; ++kh)
        loadA16x32(Ka[jt][kh], k32 + (size_t)(j0 + 16 * jt + ln) * 32 + 4 * h + 16 * kh);

    // ---- S^T tiles: s[q][jt] ----------------------------------------------
    v8f s[2][2];
#pragma unroll
    for (int q = 0; q < 2; ++q)
#pragma unroll
      for (int jt = 0; jt < 2; ++jt) {
        v8f t0 = __builtin_amdgcn_wmma_f32_16x16x32_bf16(false, Ka[jt][0].v, false, qb[q][0].v,
                                                         (short)0, zero, false, false);
        s[q][jt] = __builtin_amdgcn_wmma_f32_16x16x32_bf16(false, Ka[jt][1].v, false, qb[q][1].v,
                                                           (short)0, t0, false, false);
      }

    // ---- online softmax + P^T repack per query subtile --------------------
    FragBF Pb[2];
#pragma unroll
    for (int q = 0; q < 2; ++q) {
      float tmax = -1e30f;
#pragma unroll
      for (int e = 0; e < 8; ++e) tmax = fmaxf(tmax, fmaxf(s[q][0][e], s[q][1][e]));
      tmax = fmaxf(tmax, __shfl_xor(tmax, 16, 32));
      const float nm = fmaxf(m_run[q], tmax);
      const float corr = __expf(m_run[q] - nm);
      float p0[8], p1[8], ls = 0.0f;
#pragma unroll
      for (int e = 0; e < 8; ++e) {
        p0[e] = __expf(s[q][0][e] - nm);
        p1[e] = __expf(s[q][1][e] - nm);
        ls += p0[e] + p1[e];
      }
      ls += __shfl_xor(ls, 16, 32);
      l_run[q] = l_run[q] * corr + ls;
      m_run[q] = nm;
#pragma unroll
      for (int t = 0; t < 4; ++t)
#pragma unroll
        for (int e = 0; e < 8; ++e) acc[q][t][e] *= corr;

      // P^T (C/D layout) -> B-operand fragment. B VGPR r needs j = 2(r&3)+8h_own+16(r>>2);
      // source: tile (r>>2), D-VGPR (2r)&7, own lane-half.
      float same[8], cross[8];
#pragma unroll
      for (int e = 0; e < 8; ++e) {
        same[e]  = h ? p1[e] : p0[e];
        cross[e] = __shfl_xor(h ? p0[e] : p1[e], 16, 32);
      }
#pragma unroll
      for (int r = 0; r < 8; ++r) {
        const int hs = r >> 2;  // compile-time source half
        const float v0 = hs ? p1[(2 * r) & 7]     : p0[(2 * r) & 7];
        const float v1 = hs ? p1[(2 * r + 1) & 7] : p0[(2 * r + 1) & 7];
        Pb[q].u[r] = cvt_pk_bf16(v0, v1);
      }
    }

    // ---- O^T += V_slice^T (16c x 32j) @ P^T, V frags shared by both q -----
#pragma unroll
    for (int t = 0; t < 4; ++t) {
      FragBF Va;
      loadA16x32(Va, v32 + (size_t)(w * 64 + t * 16 + ln) * 1024 + (j0 >> 1) + 4 * h);
#pragma unroll
      for (int q = 0; q < 2; ++q)
        acc[q][t] = __builtin_amdgcn_wmma_f32_16x16x32_bf16(false, Va.v, false, Pb[q].v,
                                                            (short)0, acc[q][t], false, false);
    }
  }

  // ---- epilogue: out = gamma * O / l + x ----------------------------------
#pragma unroll
  for (int q = 0; q < 2; ++q) {
    const float inv = 1.0f / l_run[q];
#pragma unroll
    for (int t = 0; t < 4; ++t)
#pragma unroll
      for (int e = 0; e < 8; ++e) {
        const int c = w * 64 + t * 16 + e + 8 * h;
        const size_t idx = ((size_t)(b * C_ + c)) * L_ + l0 + 16 * q + ln;
        out[idx] = gamma * acc[q][t][e] * inv + x[idx];
      }
  }
}

// ---------------------------------------------------------------------------
extern "C" void kernel_launch(void* const* d_in, const int* in_sizes, int n_in,
                              void* d_out, int out_size, void* d_ws, size_t ws_size,
                              hipStream_t stream)
{
  (void)in_sizes; (void)n_in; (void)out_size; (void)ws_size;
  const float* x     = (const float*)d_in[0];
  const float* Wq    = (const float*)d_in[1];
  const float* bq    = (const float*)d_in[2];
  const float* Wk    = (const float*)d_in[3];
  const float* bk    = (const float*)d_in[4];
  const float* Wv    = (const float*)d_in[5];
  const float* bv    = (const float*)d_in[6];
  const float* gamma = (const float*)d_in[7];
  float* out = (float*)d_out;

  unsigned short* ws = (unsigned short*)d_ws;
  const size_t qN = (size_t)B_ * L_ * D_;        // 1,048,576 bf16 elems
  unsigned short* qbuf = ws;                     // [B][L][64]
  unsigned short* kbuf = ws + qN;                // [B][L][64] (= k^T)
  unsigned short* vbuf = ws + 2 * qN;            // [B][C][L], total ws ~20 MB

  proj_bf16_kernel<1><<<512,  256, 0, stream>>>(x, Wq, bq, qbuf,  64,  4);
  proj_bf16_kernel<1><<<512,  256, 0, stream>>>(x, Wk, bk, kbuf,  64,  4);
  proj_bf16_kernel<0><<<4096, 256, 0, stream>>>(x, Wv, bv, vbuf, 512, 32);
  attn_kernel<<<B_ * (L_ / 32), 256, 0, stream>>>(qbuf, kbuf, vbuf, x, gamma, out);
}
